// SparseAttention_35261681500425
// MI455X (gfx1250) — matrix operand order repaired and checked by reading, not compiled, not run on hardware
//
#include <hip/hip_runtime.h>
#include <math.h>

typedef __attribute__((ext_vector_type(16))) _Float16 v16h;
typedef __attribute__((ext_vector_type(8)))  _Float16 v8h;
typedef __attribute__((ext_vector_type(2)))  _Float16 h2;
typedef __attribute__((ext_vector_type(8)))  float    v8f;
typedef __attribute__((ext_vector_type(4)))  float    v4f;

typedef unsigned int u32x4 __attribute__((ext_vector_type(4)));
typedef int          i32x4 __attribute__((ext_vector_type(4)));
typedef int          i32x8 __attribute__((ext_vector_type(8)));

namespace {
constexpr int Bc = 4;      // batch
constexpr int Lc = 2048;   // seq len
constexpr int Hc = 8;      // heads
constexpr int Ec = 64;     // qk dim
constexpr int Dc = 64;     // v dim
constexpr int Mb = 128;    // query rows per block (8 waves x 16)
constexpr int Sb = 64;     // keys per iteration
}

// Issue a TDM 2-D tile load: 64x64 f32 tile, row stride Hc*Ec elements,
// from global `gptr` into LDS byte offset `ldsOff`.  (ISA 08_async_tensor §8)
__device__ __forceinline__ void tdm_load_tile(const float* gptr, unsigned ldsOff)
{
    const unsigned long long ga = (unsigned long long)(uintptr_t)gptr;
    u32x4 g0;
    g0[0] = 1u;                                            // count=1, user mode
    g0[1] = ldsOff;                                        // lds_addr (bytes)
    g0[2] = (unsigned)(ga & 0xFFFFFFFFu);                  // global_addr[31:0]
    g0[3] = (unsigned)((ga >> 32) & 0x1FFFFFFu) | (2u << 30); // addr[56:32] | type=2
    i32x8 g1;
    g1[0] = (int)(2u << 16);        // workgroup_mask=0, data_size=2 (4 bytes)
    g1[1] = (int)(64u << 16);       // tensor_dim0 = 64 (bits 63:48)
    g1[2] = (int)(2048u << 16);     // tensor_dim0 hi=0, tensor_dim1 = 2048 (bits 95:80)
    g1[3] = (int)(64u << 16);       // tensor_dim1 hi=0, tile_dim0 = 64 (bits 127:112)
    g1[4] = 64;                     // tile_dim1 = 64, tile_dim2 = 0
    g1[5] = (int)(Hc * Ec);         // tensor_dim0_stride = 512 (bits 191:160)
    g1[6] = 0;                      // stride hi, tensor_dim1_stride lo = 0
    g1[7] = 0;
    i32x4 g2 = {0, 0, 0, 0};        // 2-D tensor: groups 2/3 unused
    i32x4 g3 = {0, 0, 0, 0};
#if defined(__clang_major__) && (__clang_major__ >= 23)
    i32x8 g4 = {0, 0, 0, 0, 0, 0, 0, 0};
    __builtin_amdgcn_tensor_load_to_lds(g0, g1, g2, g3, g4, 0);
#else
    __builtin_amdgcn_tensor_load_to_lds(g0, g1, g2, g3, 0);
#endif
}

__global__ __launch_bounds__(256)
void sparse_attn_wmma(const float* __restrict__ Q, const float* __restrict__ K,
                      const float* __restrict__ V, float* __restrict__ O)
{
    __shared__ float    sKf32[2][Sb * Ec];  // 32KB TDM staging (double buffered)
    __shared__ float    sVf32[2][Sb * Ec];  // 32KB
    __shared__ _Float16 sK [Sb * Ec];       // 8KB  [key][e] f16
    __shared__ _Float16 sVt[Dc * Sb];       // 8KB  [d][key] f16 (V transposed)
    __shared__ _Float16 sP [8][16 * Sb];    // 16KB per-wave P staging

    const int tid  = threadIdx.x;
    const int lane = tid & 31;
    const int w    = tid >> 5;
    const int n    = lane & 15;              // C/B column within 16
    const int hi8  = (lane & 16) ? 8 : 0;    // C row offset / A K sub-offset
    const int klo  = (lane & 16) ? 16 : 0;   // unused; B operands use hi8 like A

    const int m0  = blockIdx.x * Mb;
    const int h   = blockIdx.y;
    const int b   = blockIdx.z;
    const int qr0 = m0 + w * 16;
    const int qi  = qr0 + n;

    const float cs = 0.125f * 1.4426950408889634f;   // scale * log2(e), pre-folded

    const float* Kbh = K + (((size_t)b * Lc) * Hc + h) * (size_t)Ec;
    const float* Vbh = V + (((size_t)b * Lc) * Hc + h) * (size_t)Ec;

    // LDS byte offsets (low 32 bits of flat LDS address == LDS offset, ISA §10.2)
    const unsigned offK[2] = { (unsigned)(uintptr_t)&sKf32[0][0],
                               (unsigned)(uintptr_t)&sKf32[1][0] };
    const unsigned offV[2] = { (unsigned)(uintptr_t)&sVf32[0][0],
                               (unsigned)(uintptr_t)&sVf32[1][0] };

    // ---- Q rows as two A operands (e-chunks 0..31, 32..63) ----
    const float* qrow = Q + (((size_t)b * Lc + qi) * Hc + h) * (size_t)Ec;
    v16h aQ0, aQ1;
    #pragma unroll
    for (int m = 0; m < 8; ++m) {
        aQ0[m]     = (_Float16)qrow[hi8 + m];
        aQ0[m + 8] = (_Float16)qrow[16 + hi8 + m];
        aQ1[m]     = (_Float16)qrow[32 + hi8 + m];
        aQ1[m + 8] = (_Float16)qrow[48 + hi8 + m];
    }
    v16h ones;
    #pragma unroll
    for (int m = 0; m < 16; ++m) ones[m] = (_Float16)1.0f;

    v8f acc[5];                               // 4 output tiles + row-sum tile
    #pragma unroll
    for (int t = 0; t < 5; ++t)
        #pragma unroll
        for (int r = 0; r < 8; ++r) acc[t][r] = 0.0f;

    float mrow[8];
    #pragma unroll
    for (int r = 0; r < 8; ++r) mrow[r] = -__builtin_inff();

    // conversion-pass ownership: thread owns keys {2*kp, 2*kp+1}, e in [8*e8, 8*e8+8)
    const int kp   = tid & 31;
    const int e8   = tid >> 5;
    const int key0 = kp * 2;
    const int eb   = e8 * 8;

    const int ktmax = (m0 + Mb - 1) >> 6;     // inclusive last 64-key tile (causal)

    // ---- prologue: DMA tile 0 into staging buffer 0 (wave 0 only; EXEC ignored) ----
    if (w == 0) {
        tdm_load_tile(Kbh, offK[0]);
        tdm_load_tile(Vbh, offV[0]);
    }

    for (int kt = 0; kt <= ktmax; ++kt) {
        const int s0  = kt * Sb;
        const int buf = kt & 1;

        if (w == 0) {
            if (kt < ktmax) {                 // issue next tile, then wait for current
                tdm_load_tile(Kbh + (size_t)(s0 + Sb) * Hc * Ec, offK[buf ^ 1]);
                tdm_load_tile(Vbh + (size_t)(s0 + Sb) * Hc * Ec, offV[buf ^ 1]);
                __builtin_amdgcn_s_wait_tensorcnt(2);   // current tile's 2 ops done
            } else {
                __builtin_amdgcn_s_wait_tensorcnt(0);
            }
        }
        __syncthreads();                       // staging[buf] visible to all waves

        // ---- block-wide f32 -> f16 conversion pass (LDS -> LDS) ----
        {
            const float* bK = &sKf32[buf][0];
            const float* bV = &sVf32[buf][0];
            v4f k0a = *(const v4f*)&bK[(key0    ) * Ec + eb];
            v4f k0b = *(const v4f*)&bK[(key0    ) * Ec + eb + 4];
            v4f k1a = *(const v4f*)&bK[(key0 + 1) * Ec + eb];
            v4f k1b = *(const v4f*)&bK[(key0 + 1) * Ec + eb + 4];
            v4f v0a = *(const v4f*)&bV[(key0    ) * Ec + eb];
            v4f v0b = *(const v4f*)&bV[(key0    ) * Ec + eb + 4];
            v4f v1a = *(const v4f*)&bV[(key0 + 1) * Ec + eb];
            v4f v1b = *(const v4f*)&bV[(key0 + 1) * Ec + eb + 4];
            v8h hk0, hk1;
            #pragma unroll
            for (int m = 0; m < 4; ++m) {
                hk0[m] = (_Float16)k0a[m]; hk0[m + 4] = (_Float16)k0b[m];
                hk1[m] = (_Float16)k1a[m]; hk1[m + 4] = (_Float16)k1b[m];
            }
            *(v8h*)&sK[(key0    ) * Ec + eb] = hk0;       // ds_store_b128
            *(v8h*)&sK[(key0 + 1) * Ec + eb] = hk1;
            #pragma unroll
            for (int m = 0; m < 8; ++m) {                 // V transpose, paired keys
                h2 pv;
                pv[0] = (_Float16)(m < 4 ? v0a[m] : v0b[m - 4]);
                pv[1] = (_Float16)(m < 4 ? v1a[m] : v1b[m - 4]);
                *(h2*)&sVt[(eb + m) * Sb + key0] = pv;    // ds_store_b32
            }
        }
        asm volatile("s_wait_dscnt 0" ::: "memory");       // reads done before signal
        __syncthreads();                                   // f16 tiles ready

        if (s0 > qr0 + 15) continue;           // tile fully above diagonal for this wave

        // ---- scores: S(16x64) = Q(16x64) @ K^T(64x64), four 16x16 C tiles ----
        v8f c[4];
        #pragma unroll
        for (int t = 0; t < 4; ++t) {
            const _Float16* kr = &sK[(t * 16 + n) * Ec];
            v16h b0, b1;
            #pragma unroll
            for (int m = 0; m < 16; ++m) { b0[m] = kr[hi8 + m + (m & 8)]; b1[m] = kr[32 + hi8 + m + (m & 8)]; }
            v8f z;
            #pragma unroll
            for (int r = 0; r < 8; ++r) z[r] = 0.0f;
            z    = __builtin_amdgcn_wmma_f32_16x16x32_f16(false, aQ0, false, b0, (short)0, z, false, false);
            c[t] = __builtin_amdgcn_wmma_f32_16x16x32_f16(false, aQ1, false, b1, (short)0, z, false, false);
        }

        // ---- wave-uniform tile classification (SALU) ----
        const bool needCausal = (s0 + Sb - 1) > qr0;
        const int  dmaxT = qr0 + 15 - s0;
        const int  dminT = qr0 - (s0 + Sb - 1);
        bool needPow2 = false;
        if (dmaxT >= 2) {
            const int lo2 = dminT < 2 ? 2 : dminT;
            const int np  = (lo2 <= 2) ? 2 : (1 << (32 - __builtin_clz(lo2 - 1)));
            needPow2 = (np <= dmaxT);
        }

        // ---- masking + scale (into log2-scaled domain) ----
        if (!needCausal && !needPow2) {
            #pragma unroll
            for (int t = 0; t < 4; ++t)
                #pragma unroll
                for (int r = 0; r < 8; ++r) c[t][r] *= cs;
        } else if (!needCausal) {
            #pragma unroll
            for (int t = 0; t < 4; ++t) {
                const int gj = s0 + t * 16 + n;
                #pragma unroll
                for (int r = 0; r < 8; ++r) {
                    const int d = (qr0 + hi8 + r) - gj;
                    float sl = c[t][r] * cs;
                    if (d >= 2 && (d & (d - 1)) == 0) sl = 0.0f;   // strided value mask
                    c[t][r] = sl;
                }
            }
        } else {
            #pragma unroll
            for (int t = 0; t < 4; ++t) {
                const int gj = s0 + t * 16 + n;
                #pragma unroll
                for (int r = 0; r < 8; ++r) {
                    const int d = (qr0 + hi8 + r) - gj;
                    float sl = c[t][r] * cs;
                    if (d >= 2 && (d & (d - 1)) == 0) sl = 0.0f;
                    if (d < 0) sl = -__builtin_inff();             // causal
                    c[t][r] = sl;
                }
            }
        }

        // ---- row max: packed-f16 shuffle reduction over the 16-lane group ----
        float tmax[8];
        #pragma unroll
        for (int r = 0; r < 8; ++r)
            tmax[r] = fmaxf(fmaxf(c[0][r], c[1][r]), fmaxf(c[2][r], c[3][r]));
        h2 hm[4];
        #pragma unroll
        for (int i = 0; i < 4; ++i) {
            hm[i][0] = (_Float16)tmax[2 * i];
            hm[i][1] = (_Float16)tmax[2 * i + 1];
        }
        #pragma unroll
        for (int msk = 1; msk < 16; msk <<= 1) {
            #pragma unroll
            for (int i = 0; i < 4; ++i) {
                int o = __shfl_xor(__builtin_bit_cast(int, hm[i]), msk, 32);
                hm[i] = __builtin_elementwise_max(hm[i], __builtin_bit_cast(h2, o));
            }
        }
        #pragma unroll
        for (int r = 0; r < 8; ++r) tmax[r] = (float)hm[r >> 1][r & 1];

        // ---- rescale only when some row's max actually grew (ballot-gated) ----
        bool upd = false;
        #pragma unroll
        for (int r = 0; r < 8; ++r) upd = upd || (tmax[r] > mrow[r]);
        if (__any(upd)) {
            #pragma unroll
            for (int r = 0; r < 8; ++r) {
                const float mnew = fmaxf(mrow[r], tmax[r]);
                const float a    = exp2f(mrow[r] - mnew);
                mrow[r] = mnew;
                #pragma unroll
                for (int t = 0; t < 5; ++t) acc[t][r] *= a;
            }
        }

        // ---- P = exp2(sl - m), staged to LDS for C->A relayout ----
        _Float16* sp = sP[w];
        #pragma unroll
        for (int t = 0; t < 4; ++t)
            #pragma unroll
            for (int r = 0; r < 8; ++r)
                sp[(hi8 + r) * Sb + t * 16 + n] = (_Float16)exp2f(c[t][r] - mrow[r]);
        asm volatile("s_wait_dscnt 0" ::: "memory");    // same-wave DS W->R ordering
        v16h aP0, aP1;
        {
            const _Float16* pr = &sp[n * Sb];
            #pragma unroll
            for (int m = 0; m < 8; ++m) {
                aP0[m]     = pr[hi8 + m];
                aP0[m + 8] = pr[16 + hi8 + m];
                aP1[m]     = pr[32 + hi8 + m];
                aP1[m + 8] = pr[48 + hi8 + m];
            }
        }

        // ---- O(16x64) += P(16x64) @ V(64x64); row-sum fused as a 5th tile ----
        #pragma unroll
        for (int t = 0; t < 4; ++t) {
            const _Float16* vc = &sVt[(t * 16 + n) * Sb];
            v16h b0, b1;
            #pragma unroll
            for (int m = 0; m < 16; ++m) { b0[m] = vc[hi8 + m + (m & 8)]; b1[m] = vc[32 + hi8 + m + (m & 8)]; }
            acc[t] = __builtin_amdgcn_wmma_f32_16x16x32_f16(false, aP0, false, b0, (short)0, acc[t], false, false);
            acc[t] = __builtin_amdgcn_wmma_f32_16x16x32_f16(false, aP1, false, b1, (short)0, acc[t], false, false);
        }
        acc[4] = __builtin_amdgcn_wmma_f32_16x16x32_f16(false, aP0, false, ones, (short)0, acc[4], false, false);
        acc[4] = __builtin_amdgcn_wmma_f32_16x16x32_f16(false, aP1, false, ones, (short)0, acc[4], false, false);
    }

    // ---- epilogue: normalize by fused row-sum and store (B, L, H, D) ----
    float* ob = O + (((size_t)b * Lc + qr0) * Hc + h) * (size_t)Dc;
    #pragma unroll
    for (int r = 0; r < 8; ++r) {
        const float inv  = 1.0f / acc[4][r];
        const size_t ro  = (size_t)(hi8 + r) * Hc * Dc;
        #pragma unroll
        for (int t = 0; t < 4; ++t)
            ob[ro + t * 16 + n] = acc[t][r] * inv;
    }
}

extern "C" void kernel_launch(void* const* d_in, const int* in_sizes, int n_in,
                              void* d_out, int out_size, void* d_ws, size_t ws_size,
                              hipStream_t stream) {
    const float* Q = (const float*)d_in[0];
    const float* K = (const float*)d_in[1];
    const float* V = (const float*)d_in[2];
    float* O = (float*)d_out;
    dim3 grid(Lc / Mb, Hc, Bc);
    dim3 block(256);
    sparse_attn_wmma<<<grid, block, 0, stream>>>(Q, K, V, O);
}
